// GraphEncoder_30253749633096
// MI455X (gfx1250) — compile-verified, matrix operand order checked by reading, not run on hardware
//
#include <hip/hip_runtime.h>
#include <hip/hip_bf16.h>

typedef __attribute__((ext_vector_type(16))) _Float16 v16h;
typedef __attribute__((ext_vector_type(8)))  float    v8f;

#define LATENT 128

// ---------------------------------------------------------------- zero
__global__ void ge_zero_kernel(float* __restrict__ p, long long n) {
  long long i = (long long)blockIdx.x * blockDim.x + threadIdx.x;
  if (i < n) p[i] = 0.0f;
}

// ---------------------------------------------------------------- degree
__global__ void ge_degree_kernel(const int* __restrict__ tgt,
                                 float* __restrict__ deg, int E) {
  int e = blockIdx.x * blockDim.x + threadIdx.x;
  if (e < E) atomicAdd(&deg[tgt[e]], 1.0f);
}

// ---------------------------------------------------------------- scatter
// One wave (32 lanes) per edge; each lane moves 4 consecutive features
// via a 128-bit gather + 4 f32 atomic adds (L2-resident accumulator).
__global__ void ge_scatter_kernel(const float* __restrict__ v,
                                  const int* __restrict__ src,
                                  const int* __restrict__ tgt,
                                  const float* __restrict__ ew,
                                  float* __restrict__ aggr, int E) {
  long long gid = (long long)blockIdx.x * blockDim.x + threadIdx.x;
  int e    = (int)(gid >> 5);
  int lane = (int)(gid & 31);
  if (e >= E) return;
  int s = src[e];
  int t = tgt[e];
  float w = ew[e];
  const float4* vp = (const float4*)(v + (size_t)s * LATENT);
  float4 m = vp[lane];
  float* ap = aggr + (size_t)t * LATENT + lane * 4;
  atomicAdd(ap + 0, m.x * w);
  atomicAdd(ap + 1, m.y * w);
  atomicAdd(ap + 2, m.z * w);
  atomicAdd(ap + 3, m.w * w);
}

// ---------------------------------------------------------------- WMMA linear
// out[m][n] = vin[m][n] + relu( (aggr[m][:]/max(deg[m],1)) . W[n][:] + b[n] )
// One wave per 16x16 output tile; K=128 in 4 x v_wmma_f32_16x16x32_f16.
__global__ void ge_wmma_linear_kernel(const float* __restrict__ aggr,
                                      const float* __restrict__ deg,
                                      const float* __restrict__ vin,
                                      const float* __restrict__ W,
                                      const float* __restrict__ b,
                                      float* __restrict__ out, int M) {
  const int lane          = threadIdx.x & 31;
  const int waveInBlock   = threadIdx.x >> 5;
  const int wavesPerBlock = blockDim.x >> 5;
  const int tile  = blockIdx.x * wavesPerBlock + waveInBlock;
  const int nTilesN = LATENT / 16;                // 8
  const int mTile = tile / nTilesN;
  const int nTile = tile % nTilesN;
  const int m0 = mTile * 16;
  const int n0 = nTile * 16;
  if (m0 >= M) return;                            // wave-uniform guard: EXEC stays all-1s

  const int half = lane >> 4;                     // 0 or 1
  const int l15  = lane & 15;
  const int rowA = m0 + l15;                      // A-matrix row held by this lane
  const int colB = n0 + l15;                      // B-matrix column held by this lane

  const float invd = 1.0f / fmaxf(deg[rowA], 1.0f);
  const float* __restrict__ arow = aggr + (size_t)rowA * LATENT;
  const float* __restrict__ wrow = W    + (size_t)colB * LATENT;   // B[k][n] = W[n][k]

  v8f acc = {};
  #pragma unroll
  for (int c = 0; c < 4; ++c) {
    const int kb = c * 32 + half * 8;             // lane-half K offset (ISA 16-bit A layout)
    v16h af, bf;
    #pragma unroll
    for (int h = 0; h < 16; ++h) {
      const int k = kb + (h & 7) + ((h >> 3) << 4);   // VGPR0-3: K=kb..kb+7, VGPR4-7: K=kb+16..kb+23
      af[h] = (_Float16)(arow[k] * invd);
      bf[h] = (_Float16)(wrow[k]);
    }
    // D = A x B + C  (8 args: neg_a, A, neg_b, B, c_mod, C, reuse_a, reuse_b)
    acc = __builtin_amdgcn_wmma_f32_16x16x32_f16(false, af, false, bf,
                                                 (short)0, acc, false, false);
  }

  const float bias = b[colB];
  #pragma unroll
  for (int r = 0; r < 8; ++r) {                   // C/D layout: VGPR r -> M = m0 + r + 8*half
    const int m = m0 + r + half * 8;
    float val = acc[r] + bias;
    val = val > 0.0f ? val : 0.0f;
    const size_t idx = (size_t)m * LATENT + colB;
    out[idx] = vin[idx] + val;                    // residual; same thread reads+writes idx
  }
}

// ---------------------------------------------------------------- launch
extern "C" void kernel_launch(void* const* d_in, const int* in_sizes, int n_in,
                              void* d_out, int out_size, void* d_ws, size_t ws_size,
                              hipStream_t stream) {
  const int   E   = in_sizes[0] / 2;
  const int*  src = (const int*)d_in[0];
  const int*  tgt = src + E;
  const float* ew = (const float*)d_in[1];
  const float* ve = (const float*)d_in[2];
  const int    M  = in_sizes[2] / LATENT;
  const float* W1 = (const float*)d_in[3];
  const float* b1 = (const float*)d_in[4];
  const float* W2 = (const float*)d_in[5];
  const float* b2 = (const float*)d_in[6];
  float* out = (float*)d_out;

  float* aggr = (float*)d_ws;                     // M*128 f32
  float* deg  = aggr + (size_t)M * LATENT;        // M f32

  const int TB = 256;

  // zero accumulator + degrees
  long long zn = (long long)M * LATENT + M;
  ge_zero_kernel<<<(unsigned)((zn + TB - 1) / TB), TB, 0, stream>>>(aggr, zn);

  // degrees (shared by both layers)
  ge_degree_kernel<<<(E + TB - 1) / TB, TB, 0, stream>>>(tgt, deg, E);

  const long long sThreads = (long long)E * 32;
  const unsigned  sBlocks  = (unsigned)((sThreads + TB - 1) / TB);

  const int mTiles = (M + 15) / 16;
  const int waves  = mTiles * (LATENT / 16);
  const int wBlocks = (waves + (TB / 32) - 1) / (TB / 32);

  // ---- layer 1
  ge_scatter_kernel<<<sBlocks, TB, 0, stream>>>(ve, src, tgt, ew, aggr, E);
  ge_wmma_linear_kernel<<<wBlocks, TB, 0, stream>>>(aggr, deg, ve, W1, b1, out, M);

  // ---- layer 2 (re-zero accumulator, message-pass on layer-1 output)
  long long an = (long long)M * LATENT;
  ge_zero_kernel<<<(unsigned)((an + TB - 1) / TB), TB, 0, stream>>>(aggr, an);
  ge_scatter_kernel<<<sBlocks, TB, 0, stream>>>(out, src, tgt, ew, aggr, E);
  ge_wmma_linear_kernel<<<wBlocks, TB, 0, stream>>>(aggr, deg, out, W2, b2, out, M);
}
